// FilteredLeakyReLU_30494267802249
// MI455X (gfx1250) — compile-verified, matrix operand order checked by reading, not run on hardware
//
#include <hip/hip_runtime.h>
#include <hip/hip_bf16.h>

typedef __attribute__((ext_vector_type(16))) _Float16 v16h;
typedef __attribute__((ext_vector_type(8)))  _Float16 v8h;
typedef __attribute__((ext_vector_type(8)))  float    v8f;

// x: (8*128, 128, 128) fp32; f: 12 taps. Per 16x16 output tile:
//   input halo [16t-5, 16t+20] (26 vals, padded to 32); intermediate local [0,41] (padded 48).
// Up (gain 2/pass): y[i] = 2*sum_{s=0..5} f[2s+(i&1)] * x[(i>>1)+5-s]   (local)
// Down:             o[m] =   sum_{t=0..11} f[t]       * z[2m+11-t]      (local)
//
// Dataflow (data always WMMA A-operand, filters always B; every LDS store is a
// contiguous transposed b128, every fragment load is 2x b128):
//   XT[q][p]  --A-->  D=Xt*AuT  --st_tr-->  Y1[i][q]
//   Y1[i][q]  --A-->  D=Y1*Bu   --st_tr+lrelu--> ZT[j][i]
//   ZT[j][i]  --A-->  D=Zt*AdT  --st_tr-->  TL[o][j]
//   TL[o][j]  --A-->  D=T*Bd    --> global out
// Constant matrices (baked once to d_ws by prep kernel, f16):
//   UF[rc][k] = 2*f[2s+(rc&1)], s=(rc>>1)+5-k, rc<42   (48x32)  B-frags ub0..ub2
//   DF[rc][k] = f[2rc+11-k]                            (16x64)  B-frags db0,db1

constexpr int HW = 128;
constexpr int PX = 40;   // XT pitch (halves), 32 rows (q)
constexpr int PY = 40;   // Y1 pitch, 48 rows (i)
constexpr int PZ = 72;   // ZT pitch, 48 rows (j), 64 cols (i) + pad
constexpr int PT = 72;   // TL pitch, 16 rows (o), 64 cols (j) + pad
constexpr int UF_ELEMS = 48 * 32;
constexpr int DF_ELEMS = 16 * 64;

#define WMMA_F16(A, B, C) \
  __builtin_amdgcn_wmma_f32_16x16x32_f16(false, (A), false, (B), (short)0, (C), false, false)

// Fragment load from row-major L[r][k] (pitch % 8 == 0, k0 in {0,32}): 2 aligned b128.
// Per-lane layout: lane&15 selects row, halves hold k = k0 + (h&7) + ((h>>3)<<4) + (lane>=16?8:0).
__device__ __forceinline__ v16h ld_frag(const _Float16* L, int pitch, int lane, int r0, int k0) {
  const int r = r0 + (lane & 15);
  const int kh = (lane & 16) ? 8 : 0;
  const _Float16* p = L + r * pitch + k0 + kh;
  const v8h lo = *(const v8h*)p;
  const v8h hi = *(const v8h*)(p + 16);
  return __builtin_shufflevector(lo, hi, 0, 1, 2, 3, 4, 5, 6, 7, 8, 9, 10, 11, 12, 13, 14, 15);
}

// Store D accumulator transposed: L[n][m] (reg r -> m = m0+hi+r contiguous) -> one b128.
template <bool LRELU>
__device__ __forceinline__ void st_tr(_Float16* L, int pitch, int lane, int n0, int m0, v8f a) {
  const int n = n0 + (lane & 15);
  const int m = m0 + ((lane & 16) ? 8 : 0);
  v8h v;
#pragma unroll
  for (int r = 0; r < 8; ++r) {
    float t = a[r];
    if (LRELU) t = (t >= 0.0f) ? t : t * 0.01f;
    v[r] = (_Float16)t;
  }
  *(v8h*)(L + n * pitch + m) = v;
}

__device__ __forceinline__ void lds_fence() {
  asm volatile("s_wait_dscnt 0" ::: "memory");
}

// ---- prep: bake banded filter matrices to f16 in workspace (runs once, tiny) ----
__global__ void flrelu_prep_kernel(const float* __restrict__ f, _Float16* __restrict__ ws) {
  const int t = blockIdx.x * blockDim.x + threadIdx.x;
  if (t < UF_ELEMS) {
    const int rc = t >> 5, k = t & 31;
    const int s = (rc >> 1) + 5 - k;
    const float v = (rc < 42 && s >= 0 && s < 6) ? 2.0f * f[2 * s + (rc & 1)] : 0.0f;
    ws[t] = (_Float16)v;
  } else if (t < UF_ELEMS + DF_ELEMS) {
    const int u = t - UF_ELEMS;
    const int rc = u >> 6, k = u & 63;
    const int tt = 2 * rc + 11 - k;
    const float v = (tt >= 0 && tt < 12) ? f[tt] : 0.0f;
    ws[t] = (_Float16)v;
  }
}

__global__ __launch_bounds__(32) void flrelu_wmma_kernel(const float* __restrict__ x,
                                                         const _Float16* __restrict__ ws,
                                                         float* __restrict__ out) {
  __shared__ __align__(16) _Float16 XT[32 * PX];  // XT[q][p] = x halo transposed, f16
  __shared__ __align__(16) _Float16 Y1[48 * PY];  // Y1[i][q]  after vertical up-FIR
  __shared__ __align__(16) _Float16 ZT[48 * PZ];  // ZT[j][i]  after horiz up-FIR + lrelu
  __shared__ __align__(16) _Float16 TL[16 * PT];  // TL[o][j]  after vertical down-FIR

  const int lane = threadIdx.x;
  const int tile = blockIdx.x;
  const int nc = tile >> 6;        // image 0..1023
  const int tr = (tile >> 3) & 7;  // tile row
  const int tc = tile & 7;         // tile col

  // Constant B-fragments from workspace (L2-hot, 10x b128 total).
  const _Float16* UF = ws;             // [48][32]
  const _Float16* DF = ws + UF_ELEMS;  // [16][64]
  const v16h ub0 = ld_frag(UF, 32, lane, 0, 0);
  const v16h ub1 = ld_frag(UF, 32, lane, 16, 0);
  const v16h ub2 = ld_frag(UF, 32, lane, 32, 0);
  const v16h db0 = ld_frag(DF, 64, lane, 0, 0);
  const v16h db1 = ld_frag(DF, 64, lane, 0, 32);

  // Zero K-pad columns (48..63) of ZT / TL (filter matrices are 0 there; operands
  // must just be finite).
  {
    const v8h z = {};
    for (int r = lane; r < 48; r += 32) {
      *(v8h*)(ZT + r * PZ + 48) = z;
      *(v8h*)(ZT + r * PZ + 56) = z;
    }
    if (lane < 16) {
      *(v8h*)(TL + lane * PT + 48) = z;
      *(v8h*)(TL + lane * PT + 56) = z;
    }
  }

  // ---- Stage 0: load 26x26 halo (zero-padded to 32x32) transposed into XT ----
  // Branchless: clamped-address unconditional loads + select, 4x b128 LDS stores.
  {
    const int q = lane;
    const int gc0 = tc * 16 - 5 + q;
    const int gc = min(max(gc0, 0), HW - 1);
    const bool cok = (gc0 == gc);
    const float* xb = x + (size_t)nc * HW * HW + gc;
#pragma unroll
    for (int pb = 0; pb < 32; pb += 8) {
      v8h v;
#pragma unroll
      for (int j = 0; j < 8; ++j) {
        const int gr0 = tr * 16 - 5 + pb + j;
        const int gr = min(max(gr0, 0), HW - 1);
        float t = xb[gr * HW];
        t = (cok && gr0 == gr) ? t : 0.0f;
        v[j] = (_Float16)t;
      }
      *(v8h*)(XT + q * PX + pb) = v;
    }
  }
  lds_fence();

  // ---- Stage 1: vertical up-FIR.  D[q][i] = sum_p XT[q][p] * UF[i][p]  -> Y1[i][q]
#pragma unroll
  for (int mc = 0; mc < 2; ++mc) {
    const v16h a = ld_frag(XT, PX, lane, mc * 16, 0);
    v8f d0 = {}, d1 = {}, d2 = {};
    d0 = WMMA_F16(a, ub0, d0);
    d1 = WMMA_F16(a, ub1, d1);
    d2 = WMMA_F16(a, ub2, d2);
    st_tr<false>(Y1, PY, lane, 0, mc * 16, d0);
    st_tr<false>(Y1, PY, lane, 16, mc * 16, d1);
    st_tr<false>(Y1, PY, lane, 32, mc * 16, d2);
  }
  lds_fence();

  // ---- Stage 2: horizontal up-FIR + lrelu.  D[i][j] = sum_q Y1[i][q] * UF[j][q] -> ZT[j][i]
#pragma unroll
  for (int mc = 0; mc < 3; ++mc) {
    const v16h a = ld_frag(Y1, PY, lane, mc * 16, 0);
    v8f d0 = {}, d1 = {}, d2 = {};
    d0 = WMMA_F16(a, ub0, d0);
    d1 = WMMA_F16(a, ub1, d1);
    d2 = WMMA_F16(a, ub2, d2);
    st_tr<true>(ZT, PZ, lane, 0, mc * 16, d0);
    st_tr<true>(ZT, PZ, lane, 16, mc * 16, d1);
    st_tr<true>(ZT, PZ, lane, 32, mc * 16, d2);
  }
  lds_fence();

  // ---- Stage 3: vertical down-FIR.  D[j][o] = sum_i ZT[j][i] * DF[o][i] -> TL[o][j]
#pragma unroll
  for (int mc = 0; mc < 3; ++mc) {
    v8f acc = {};
    const v16h a0 = ld_frag(ZT, PZ, lane, mc * 16, 0);
    acc = WMMA_F16(a0, db0, acc);
    const v16h a1 = ld_frag(ZT, PZ, lane, mc * 16, 32);
    acc = WMMA_F16(a1, db1, acc);
    st_tr<false>(TL, PT, lane, 0, mc * 16, acc);
  }
  lds_fence();

  // ---- Stage 4: horizontal down-FIR.  D[o][w] = sum_j TL[o][j] * DF[w][j] ----
  {
    const v16h a0 = ld_frag(TL, PT, lane, 0, 0);
    const v16h a1 = ld_frag(TL, PT, lane, 0, 32);
    v8f acc = {};
    acc = WMMA_F16(a0, db0, acc);
    acc = WMMA_F16(a1, db1, acc);

    // D layout: reg r -> row o = r + 8*hi, lane&15 -> col w.
    const int w = lane & 15;
    const int hi = (lane & 16) ? 8 : 0;
    float* ob = out + (size_t)nc * HW * HW + (size_t)(tr * 16 + hi) * HW + tc * 16 + w;
#pragma unroll
    for (int r = 0; r < 8; ++r) ob[r * HW] = acc[r];
  }
}

extern "C" void kernel_launch(void* const* d_in, const int* in_sizes, int n_in,
                              void* d_out, int out_size, void* d_ws, size_t ws_size,
                              hipStream_t stream) {
  (void)in_sizes; (void)n_in; (void)out_size; (void)ws_size;
  const float* x = (const float*)d_in[0];   // (8*128, 128, 128) fp32
  const float* f = (const float*)d_in[1];   // 12 taps fp32
  float* out = (float*)d_out;               // (8*128, 128, 128) fp32
  _Float16* ws = (_Float16*)d_ws;           // 5120 B: UF (48x32) + DF (16x64), f16

  // Bake constant banded filter matrices (idempotent, same stream -> ordered).
  flrelu_prep_kernel<<<dim3((UF_ELEMS + DF_ELEMS + 255) / 256), dim3(256), 0, stream>>>(f, ws);
  // 1024 images * 8x8 tiles of 16x16; one wave32 per tile, 1-wave workgroups.
  flrelu_wmma_kernel<<<dim3(1024 * 8 * 8), dim3(32), 0, stream>>>(x, ws, out);
}
